// GeometricUnifiedAttention_16295105921214
// MI455X (gfx1250) — matrix operand order repaired and checked by reading, not compiled, not run on hardware
//
#include <hip/hip_runtime.h>

typedef __attribute__((ext_vector_type(16))) __bf16     v16bf;
typedef __attribute__((ext_vector_type(8)))  float      v8f;
typedef __attribute__((ext_vector_type(4)))  unsigned   v4u;
typedef __attribute__((ext_vector_type(2)))  unsigned   v2u;

#define Bsz 8
#define Tsz 4096
#define Dsz 1024
#define Hn  16
#define DHc 64
#define Msz (Bsz*Tsz)   // 32768
#define Ksz Dsz         // 1024
#define Nsz Dsz         // 1024

// LDS tile geometry (ushort units): 32-wide K tiles, padded pitch 40 (80B)
// -> fragment ds_load_b128 rows hit banks 20*r mod 64, all distinct.
#define PITCH   40
#define A_ROWS  128
#define B_ROWS  256
#define A_ELEMS (A_ROWS*PITCH)             // 5120
#define BUF_ELEMS (A_ELEMS + B_ROWS*PITCH) // 15360

__device__ __forceinline__ unsigned short f2bf_rne(float f) {
    unsigned u = __builtin_bit_cast(unsigned, f);
    u += 0x7fffu + ((u >> 16) & 1u);
    return (unsigned short)(u >> 16);
}
__device__ __forceinline__ float bf2f(unsigned short h) {
    unsigned u = ((unsigned)h) << 16;
    return __builtin_bit_cast(float, u);
}

// 32-bit LDS byte offset of a generic pointer into a __shared__ array
__device__ __forceinline__ unsigned lds_addr_of(const void* p) {
    return (unsigned)(unsigned long long)
        (const __attribute__((address_space(3))) void*)p;
}
// async 16B global -> LDS copy (ASYNCcnt-tracked)
__device__ __forceinline__ void async_g2l_b128(const unsigned short* g,
                                               const unsigned short* l) {
    unsigned loff = lds_addr_of(l);
    asm volatile("global_load_async_to_lds_b128 %0, %1, off"
                 :: "v"(loff), "v"(g) : "memory");
}
__device__ __forceinline__ void wait_async0() {
    asm volatile("s_wait_asynccnt 0" ::: "memory");
}

// butterfly sum across a 16-lane row using DPP16 (pure VALU, no LDS traffic)
__device__ __forceinline__ float row16_sum(float x) {
    int t;
    t = __builtin_amdgcn_update_dpp(0, __builtin_bit_cast(int, x),
                                    0xB1, 0xf, 0xf, true);   // quad_perm(1,0,3,2)
    x += __builtin_bit_cast(float, t);
    t = __builtin_amdgcn_update_dpp(0, __builtin_bit_cast(int, x),
                                    0x4E, 0xf, 0xf, true);   // quad_perm(2,3,0,1)
    x += __builtin_bit_cast(float, t);
    t = __builtin_amdgcn_update_dpp(0, __builtin_bit_cast(int, x),
                                    0x141, 0xf, 0xf, true);  // row_half_mirror (xor4)
    x += __builtin_bit_cast(float, t);
    t = __builtin_amdgcn_update_dpp(0, __builtin_bit_cast(int, x),
                                    0x140, 0xf, 0xf, true);  // row_mirror (xor8)
    x += __builtin_bit_cast(float, t);
    return x;
}

// ---------------- fp32 -> bf16 conversion (x, Wr, Wf) ----------------
__global__ void f2bf_kernel(const float* __restrict__ in,
                            unsigned short* __restrict__ out, long n) {
    long i = (long)blockIdx.x * blockDim.x + threadIdx.x;
    long stride = (long)gridDim.x * blockDim.x;
    for (; i < n; i += stride) out[i] = f2bf_rne(in[i]);
}

// ------- fused GEMM + bias + activation -> bf16 [B,H,T,DH], LDS-staged -----
union FragAB { v16bf v; v4u q[2]; };
union FragC  { v8f   v; float f[8]; };

__global__ __launch_bounds__(256)
void gemm_act_kernel(const unsigned short* __restrict__ Xb,
                     const unsigned short* __restrict__ Wrb,
                     const unsigned short* __restrict__ Wfb,
                     const float* __restrict__ br,
                     const float* __restrict__ bfb,
                     unsigned short* __restrict__ Vs,
                     unsigned short* __restrict__ As) {
    const int proj = blockIdx.z;                       // 0: v=tanh, 1: alpha=sigmoid
    const unsigned short* __restrict__ Wb = proj ? Wfb : Wrb;
    const float* __restrict__ bias        = proj ? bfb : br;
    unsigned short* __restrict__ Y        = proj ? As  : Vs;

    const int lane  = threadIdx.x & 31;
    const int wave  = threadIdx.x >> 5;
    const int wm    = wave >> 2;                       // 0..1
    const int wn    = wave & 3;                        // 0..3
    const int tileM = blockIdx.y * 128 + wm * 64;
    const int tileN = blockIdx.x * 256 + wn * 64;
    const int half  = lane >> 4;
    const int l15   = lane & 15;

    __shared__ __align__(16) unsigned short lds[2 * BUF_ELEMS]; // 60 KB

    const unsigned short* Asrc0 = Xb + (size_t)blockIdx.y * 128 * Ksz;
    const unsigned short* Bsrc0 = Wb + (size_t)blockIdx.x * 256 * Ksz;

    auto stage = [&](int kk, int buf) {
        unsigned short* dst = &lds[buf * BUF_ELEMS];
#pragma unroll
        for (int j = 0; j < 2; ++j) {                  // A: 512 x 16B chunks
            int c = threadIdx.x + 256 * j;
            int row = c >> 2, sub = (c & 3) * 8;
            async_g2l_b128(Asrc0 + (size_t)row * Ksz + kk + sub,
                           dst + row * PITCH + sub);
        }
#pragma unroll
        for (int j = 0; j < 4; ++j) {                  // B: 1024 x 16B chunks
            int c = threadIdx.x + 256 * j;
            int row = c >> 2, sub = (c & 3) * 8;
            async_g2l_b128(Bsrc0 + (size_t)row * Ksz + kk + sub,
                           dst + A_ELEMS + row * PITCH + sub);
        }
    };

    FragC acc[4][4];
#pragma unroll
    for (int i = 0; i < 4; ++i)
#pragma unroll
        for (int j = 0; j < 4; ++j)
#pragma unroll
            for (int k = 0; k < 8; ++k) acc[i][j].f[k] = 0.0f;

    stage(0, 0);
    wait_async0();
    __syncthreads();

    int cur = 0;
    for (int kk = 0; kk < Ksz; kk += 32) {
        if (kk + 32 < Ksz) stage(kk + 32, cur ^ 1);    // overlap next copy

        const unsigned short* Lb = &lds[cur * BUF_ELEMS];
        FragAB a[4], b[4];
#pragma unroll
        for (int mi = 0; mi < 4; ++mi) {
            const unsigned short* pa =
                Lb + (wm * 64 + 16 * mi + l15) * PITCH + 8 * half;
            a[mi].q[0] = *(const v4u*)pa;
            a[mi].q[1] = *(const v4u*)(pa + 16);
        }
#pragma unroll
        for (int ni = 0; ni < 4; ++ni) {
            const unsigned short* pb =
                Lb + A_ELEMS + (wn * 64 + 16 * ni + l15) * PITCH + 8 * half;
            b[ni].q[0] = *(const v4u*)pb;
            b[ni].q[1] = *(const v4u*)(pb + 16);
        }
#pragma unroll
        for (int mi = 0; mi < 4; ++mi)
#pragma unroll
            for (int ni = 0; ni < 4; ++ni)
                acc[mi][ni].v = __builtin_amdgcn_wmma_f32_16x16x32_bf16(
                    false, a[mi].v, false, b[ni].v,
                    (short)0, acc[mi][ni].v, false, false);

        if (kk + 32 < Ksz) {
            wait_async0();
            __syncthreads();
        }
        cur ^= 1;
    }

    // Epilogue: bias + branch-light activation, write bf16 into [B,H,T,DH]
#pragma unroll
    for (int ni = 0; ni < 4; ++ni) {
        const int n  = tileN + 16 * ni + l15;
        const float bv = bias[n];
        const int h  = n >> 6;
        const int dh = n & 63;
#pragma unroll
        for (int mi = 0; mi < 4; ++mi) {
#pragma unroll
            for (int rr = 0; rr < 8; ++rr) {
                const int m  = tileM + 16 * mi + rr + 8 * half;
                const int bb = m >> 12;       // / Tsz
                const int t  = m & (Tsz - 1);
                float xv  = acc[mi][ni].f[rr] + bv;
                float arg = proj ? -xv : (-2.0f * xv);
                float e   = __expf(arg);
                float r   = __builtin_amdgcn_rcpf(1.0f + e);
                float yv  = proj ? r : fmaf(2.0f, r, -1.0f); // sigmoid : tanh
                size_t idx = (((size_t)bb * Hn + h) * Tsz + t) * DHc + dh;
                Y[idx] = f2bf_rne(yv);
            }
        }
    }
}

// ---------------- sequential RMS-gated scan over T ----------------
// One wave handles TWO (b,h) sequences: lanes 0-15 -> seq0, lanes 16-31 -> seq1.
// Each lane owns 4 channels; the DH=64 reduction is a 16-lane DPP butterfly.
__global__ __launch_bounds__(32)
void scan_kernel(const unsigned short* __restrict__ Vs,
                 const unsigned short* __restrict__ As,
                 const float* __restrict__ scale,
                 const float* __restrict__ resg,
                 float* __restrict__ out) {
    const int whalf = threadIdx.x >> 4;          // which sequence in this wave
    const int l16   = threadIdx.x & 15;
    const int bh    = blockIdx.x * 2 + whalf;    // 0..127 = b*16+h
    const int b     = bh >> 4;
    const int h     = bh & 15;
    const int dh    = l16 * 4;

    float sc[4], rg[4], st[4];
#pragma unroll
    for (int i = 0; i < 4; ++i) {
        sc[i] = scale[h * DHc + dh + i];
        rg[i] = resg[h * DHc + dh + i];
        st[i] = 0.0f;
    }

    const size_t basein = (size_t)bh * Tsz * DHc + dh;
    float* po = out + (size_t)b * Tsz * Dsz + h * DHc + dh;

    v2u pv = *(const v2u*)(Vs + basein);         // 4 bf16 v-values
    v2u pa = *(const v2u*)(As + basein);         // 4 bf16 alpha-values

    for (int t = 0; t < Tsz; ++t) {
        float v[4], a[4];
        v[0] = bf2f((unsigned short)(pv[0] & 0xffff));
        v[1] = bf2f((unsigned short)(pv[0] >> 16));
        v[2] = bf2f((unsigned short)(pv[1] & 0xffff));
        v[3] = bf2f((unsigned short)(pv[1] >> 16));
        a[0] = bf2f((unsigned short)(pa[0] & 0xffff));
        a[1] = bf2f((unsigned short)(pa[0] >> 16));
        a[2] = bf2f((unsigned short)(pa[1] & 0xffff));
        a[3] = bf2f((unsigned short)(pa[1] >> 16));
        if (t + 1 < Tsz) {                        // prefetch next step
            pv = *(const v2u*)(Vs + basein + (size_t)(t + 1) * DHc);
            pa = *(const v2u*)(As + basein + (size_t)(t + 1) * DHc);
        }
        float s[4];
#pragma unroll
        for (int i = 0; i < 4; ++i) s[i] = fmaf(a[i], st[i] - v[i], v[i]);

        float u0  = fmaf(s[1], s[1], s[0] * s[0]);
        float u1  = fmaf(s[3], s[3], s[2] * s[2]);
        float loc = row16_sum(u0 + u1);
        float rinv = __builtin_amdgcn_rsqf(fmaf(loc, 1.0f / 64.0f, 1e-12f));

        float4 o;
        float f0 = fmaf(s[0] * rinv, sc[0], v[0] * rg[0]);
        float f1 = fmaf(s[1] * rinv, sc[1], v[1] * rg[1]);
        float f2 = fmaf(s[2] * rinv, sc[2], v[2] * rg[2]);
        float f3 = fmaf(s[3] * rinv, sc[3], v[3] * rg[3]);
        st[0] = f0; st[1] = f1; st[2] = f2; st[3] = f3;
        o.x = f0; o.y = f1; o.z = f2; o.w = f3;
        *(float4*)(po + (size_t)t * Dsz) = o;     // 16B coalesced store
    }
}

extern "C" void kernel_launch(void* const* d_in, const int* in_sizes, int n_in,
                              void* d_out, int out_size, void* d_ws, size_t ws_size,
                              hipStream_t stream) {
    const float* x     = (const float*)d_in[0];
    const float* Wr    = (const float*)d_in[1];
    const float* br    = (const float*)d_in[2];
    const float* Wf    = (const float*)d_in[3];
    const float* bfb   = (const float*)d_in[4];
    const float* scale = (const float*)d_in[5];
    const float* resg  = (const float*)d_in[6];
    float* out = (float*)d_out;

    // workspace (ushort units): Xb[M*K] | Wrb[N*K] | Wfb[N*K] | Vs[M*D] | As[M*D]
    unsigned short* Xb  = (unsigned short*)d_ws;
    unsigned short* Wrb = Xb  + (size_t)Msz * Ksz;
    unsigned short* Wfb = Wrb + (size_t)Nsz * Ksz;
    unsigned short* Vs  = Wfb + (size_t)Nsz * Ksz;
    unsigned short* As  = Vs  + (size_t)Msz * Dsz;

    f2bf_kernel<<<2048, 256, 0, stream>>>(x,  Xb,  (long)Msz * Ksz);
    f2bf_kernel<<<256,  256, 0, stream>>>(Wr, Wrb, (long)Nsz * Ksz);
    f2bf_kernel<<<256,  256, 0, stream>>>(Wf, Wfb, (long)Nsz * Ksz);

    dim3 grid(Nsz / 256, Msz / 128, 2);   // (4, 256, 2)
    gemm_act_kernel<<<grid, 256, 0, stream>>>(Xb, Wrb, Wfb, br, bfb, Vs, As);

    scan_kernel<<<64, 32, 0, stream>>>(Vs, As, scale, resg, out);
}
